// ScaledDotProductAttention_42125039239929
// MI455X (gfx1250) — matrix operand order repaired and checked by reading, not compiled, not run on hardware
//
#include <hip/hip_runtime.h>
#include <hip/hip_bf16.h>

typedef __attribute__((ext_vector_type(16))) __bf16 v16bf;
typedef __attribute__((ext_vector_type(8)))  __bf16 v8bf;
typedef __attribute__((ext_vector_type(8)))  float  v8f;
typedef __attribute__((ext_vector_type(4)))  int    v4i;

#define S_LEN   2048
#define D_HEAD  64
#define NWAVES  8
#define WG_ROWS 128               /* 8 waves x 16 query rows */
#define KBLK    32                /* keys per inner block */
#define TILE_ELTS (KBLK * D_HEAD) /* 2048 floats per K or V tile */

/* ---------------- raw hardware exp2 (args are always <= 0 here) ---------------- */
#if defined(__HIP_DEVICE_COMPILE__) && __has_builtin(__builtin_amdgcn_exp2f)
#define EXP2F(x) __builtin_amdgcn_exp2f(x)
#else
#define EXP2F(x) exp2f(x)
#endif

/* ---------------- CDNA5 async global->LDS path (guarded) ---------------- */
#if __has_builtin(__builtin_amdgcn_global_load_async_to_lds_b128)
#define HAVE_ASYNC_LDS 1
#else
#define HAVE_ASYNC_LDS 0
#endif

#if HAVE_ASYNC_LDS
#if __has_builtin(__builtin_amdgcn_s_wait_asynccnt)
#define WAIT_ASYNC0() __builtin_amdgcn_s_wait_asynccnt(0)
#else
#define WAIT_ASYNC0() asm volatile("s_wait_asynccnt 0" ::: "memory")
#endif
#else
#define WAIT_ASYNC0() ((void)0)
#endif

__device__ __forceinline__ void stage_tile(const float* __restrict__ gK,
                                           const float* __restrict__ gV,
                                           float* rawK, float* rawV, int tid) {
#if HAVE_ASYNC_LDS
    /* 512 x b128 async transfers per 8KB tile; 2 per thread per tile */
    #pragma unroll
    for (int i = 0; i < TILE_ELTS / 4; i += 256) {
        const int idx = (tid + i) * 4;   /* float index, 16B granules */
        __builtin_amdgcn_global_load_async_to_lds_b128(
            (__attribute__((address_space(1))) v4i*)(gK + idx),
            (__attribute__((address_space(3))) v4i*)(rawK + idx), 0, 0);
        __builtin_amdgcn_global_load_async_to_lds_b128(
            (__attribute__((address_space(1))) v4i*)(gV + idx),
            (__attribute__((address_space(3))) v4i*)(rawV + idx), 0, 0);
    }
#else
    #pragma unroll
    for (int e = tid; e < TILE_ELTS; e += 256) {
        rawK[e] = gK[e];
        rawV[e] = gV[e];
    }
#endif
}

/* ---------------- 16-lane butterfly reductions via v_permlane16 ---------------- */
#if __has_builtin(__builtin_amdgcn_permlane16_b32)
#define PERMLN16(u, s0, s1) __builtin_amdgcn_permlane16_b32(u, u, s0, s1, false, false)
#define HAVE_PERMLANE 1
#elif __has_builtin(__builtin_amdgcn_permlane16)
#define PERMLN16(u, s0, s1) __builtin_amdgcn_permlane16(u, u, s0, s1, false, false)
#define HAVE_PERMLANE 1
#else
#define HAVE_PERMLANE 0
#endif

#if HAVE_PERMLANE
__device__ __forceinline__ float permxor(float x, unsigned s0, unsigned s1) {
    return __uint_as_float(PERMLN16(__float_as_uint(x), s0, s1));
}
__device__ __forceinline__ float row16_max(float x) {
    x = fmaxf(x, permxor(x, 0x67452301u, 0xEFCDAB89u));  /* xor 1 */
    x = fmaxf(x, permxor(x, 0x45670123u, 0xCDEF89ABu));  /* xor 3 */
    x = fmaxf(x, permxor(x, 0x32107654u, 0xBA98FEDCu));  /* xor 4 */
    x = fmaxf(x, permxor(x, 0xFEDCBA98u, 0x76543210u));  /* xor 8 */
    return x;
}
__device__ __forceinline__ float row16_sum(float x) {
    x += permxor(x, 0x67452301u, 0xEFCDAB89u);
    x += permxor(x, 0x45670123u, 0xCDEF89ABu);
    x += permxor(x, 0x32107654u, 0xBA98FEDCu);
    x += permxor(x, 0xFEDCBA98u, 0x76543210u);
    return x;
}
#else
__device__ __forceinline__ float row16_max(float x) {
    x = fmaxf(x, __shfl_xor(x, 1, 32));
    x = fmaxf(x, __shfl_xor(x, 2, 32));
    x = fmaxf(x, __shfl_xor(x, 4, 32));
    x = fmaxf(x, __shfl_xor(x, 8, 32));
    return x;
}
__device__ __forceinline__ float row16_sum(float x) {
    x += __shfl_xor(x, 1, 32);
    x += __shfl_xor(x, 2, 32);
    x += __shfl_xor(x, 4, 32);
    x += __shfl_xor(x, 8, 32);
    return x;
}
#endif

/* One 32-key block for one wave. MASKED instantiation runs only for the single
   diagonal block per wave; the interior path has no masking VALU at all.
   rsum is a per-lane PARTIAL sum (reduced across lanes only in the epilogue). */
template<bool MASKED>
__device__ __forceinline__ void attn_block(
    int kb, int q0, int half, int ln, int lane,
    const v16bf (&qa)[2], v8f (&acc)[4], float (&rmax)[8], float (&rsum)[8],
    const __bf16* __restrict__ sKt, const __bf16* __restrict__ sV,
    __bf16* __restrict__ pw)
{
    /* ---- S = (Q*scale) . K^T : two 16x16 score tiles */
    v8f s[2];
    #pragma unroll
    for (int jt = 0; jt < 2; ++jt)
        #pragma unroll
        for (int v = 0; v < 8; ++v) s[jt][v] = 0.0f;

    #pragma unroll
    for (int jt = 0; jt < 2; ++jt) {
        #pragma unroll
        for (int c = 0; c < 2; ++c) {
            /* B-layout (32x16): lane holds key column jt*16+ln; elem i -> d = 32c+8*half+i, elem 8+i -> d = 32c+16+8*half+i */
            v16bf kfrag;
            #pragma unroll
            for (int i = 0; i < 8; ++i) {
                kfrag[i]     = sKt[(32 * c + 8 * half + i) * KBLK + jt * 16 + ln];
                kfrag[8 + i] = sKt[(32 * c + 16 + 8 * half + i) * KBLK + jt * 16 + ln];
            }
            s[jt] = __builtin_amdgcn_wmma_f32_16x16x32_bf16(
                false, qa[c], false, kfrag, (short)0, s[jt], false, false);
        }
    }

    /* ---- (optional) causal mask + online softmax (base-2 domain) */
    #pragma unroll
    for (int v = 0; v < 8; ++v) {
        float s0 = s[0][v];
        float s1 = s[1][v];
        if (MASKED) {
            const int qi = q0 + v + 8 * half;
            if (kb + ln      > qi) s0 = -__builtin_inff();
            if (kb + 16 + ln > qi) s1 = -__builtin_inff();
        }

        const float m  = row16_max(fmaxf(s0, s1));
        const float nm = fmaxf(rmax[v], m);
        const float corr = EXP2F(rmax[v] - nm);   /* lane-uniform within the row */

        const float p0 = EXP2F(s0 - nm);
        const float p1 = EXP2F(s1 - nm);

        /* per-lane partial denominator; cross-lane reduction deferred to epilogue */
        rsum[v] = rsum[v] * corr + (p0 + p1);
        rmax[v] = nm;
        #pragma unroll
        for (int c = 0; c < 4; ++c) acc[c][v] *= corr;
        s[0][v] = p0; s[1][v] = p1;
    }

    /* ---- P (C-layout) -> LDS -> A-fragment layout (intra-wave) */
    #pragma unroll
    for (int v = 0; v < 8; ++v) {
        const int m = v + 8 * half;
        pw[m * KBLK + ln]      = (__bf16)s[0][v];
        pw[m * KBLK + 16 + ln] = (__bf16)s[1][v];
    }
    v8bf plo = *(const v8bf*)&pw[ln * KBLK + 8 * half];
    v8bf phi = *(const v8bf*)&pw[ln * KBLK + 16 + 8 * half];
    v16bf pa = __builtin_shufflevector(plo, phi,
                  0,1,2,3,4,5,6,7,8,9,10,11,12,13,14,15);

    /* ---- O += P . V : 4 WMMAs over D chunks */
    #pragma unroll
    for (int c = 0; c < 4; ++c) {
        v16bf vfrag;
        #pragma unroll
        for (int i = 0; i < 8; ++i) {
            vfrag[i]     = sV[(8 * half + i) * D_HEAD + 16 * c + ln];
            vfrag[8 + i] = sV[(16 + 8 * half + i) * D_HEAD + 16 * c + ln];
        }
        acc[c] = __builtin_amdgcn_wmma_f32_16x16x32_bf16(
            false, pa, false, vfrag, (short)0, acc[c], false, false);
    }
}

__global__ __launch_bounds__(256) void fa_fwd_causal_bf16wmma(
    const float* __restrict__ Qg, const float* __restrict__ Kg,
    const float* __restrict__ Vg, float* __restrict__ Og)
{
    /* double-buffered raw fp32 tiles (async destination) + bf16 compute tiles */
    __shared__ __align__(16) float  rawK[2][TILE_ELTS];          /* 16 KB */
    __shared__ __align__(16) float  rawV[2][TILE_ELTS];          /* 16 KB */
    __shared__ __align__(32) __bf16 sKt[TILE_ELTS];              /* [d][key] 4 KB */
    __shared__ __align__(32) __bf16 sV [TILE_ELTS];              /* [key][d] 4 KB */
    __shared__ __align__(32) __bf16 sP [NWAVES * 16 * KBLK];     /* 8 KB */

    const int tid  = (int)threadIdx.x;
    const int wave = tid >> 5;
    const int lane = tid & 31;
    const int half = lane >> 4;
    const int ln   = lane & 15;

    const int qb = (int)blockIdx.x * WG_ROWS;
    const int q0 = qb + wave * 16;

    const size_t base = (size_t)blockIdx.y * (size_t)(S_LEN * D_HEAD);
    const float* Qp = Qg + base;
    const float* Kp = Kg + base;
    const float* Vp = Vg + base;
    float*       Op = Og + base;

    const int nblk = (qb + WG_ROWS) / KBLK;   /* causal block count for this WG */
    /* the single diagonal (masked) block for this wave */
    const int kb_diag = ((q0 + 15) >> 5) << 5;

    /* prologue: kick off async stage of block 0 */
    stage_tile(Kp, Vp, rawK[0], rawV[0], tid);

    /* Q A-fragments (overlap global Q loads with the async K/V stream).
       A-layout (16-bit, 16x32): row m = ln; elem i -> d = 32c + 8*half + i,
       elem 8+i -> d = 32c + 16 + 8*half + i.  Pre-scaled by log2(e)/sqrt(D). */
    const float qscale = 0.125f * 1.44269504088896340736f;
    v16bf qa[2];
    #pragma unroll
    for (int c = 0; c < 2; ++c) {
        const float* qrow = Qp + (size_t)(q0 + ln) * D_HEAD + 32 * c + 8 * half;
        #pragma unroll
        for (int i = 0; i < 8; ++i) {
            qa[c][i]     = (__bf16)(qrow[i]      * qscale);
            qa[c][8 + i] = (__bf16)(qrow[16 + i] * qscale);
        }
    }

    v8f acc[4];
    #pragma unroll
    for (int c = 0; c < 4; ++c)
        #pragma unroll
        for (int v = 0; v < 8; ++v) acc[c][v] = 0.0f;

    float rmax[8], rsum[8];
    #pragma unroll
    for (int v = 0; v < 8; ++v) { rmax[v] = -__builtin_inff(); rsum[v] = 0.0f; }

    WAIT_ASYNC0();
    __syncthreads();   /* raw[0] visible to all waves */

    for (int ib = 0; ib < nblk; ++ib) {
        const int kb  = ib * KBLK;
        const int buf = ib & 1;

        /* overlap: stream next tile into the other buffer while we convert+compute */
        if (ib + 1 < nblk)
            stage_tile(Kp + (size_t)(kb + KBLK) * D_HEAD,
                       Vp + (size_t)(kb + KBLK) * D_HEAD,
                       rawK[buf ^ 1], rawV[buf ^ 1], tid);

        /* cooperative convert raw fp32 -> bf16 (K transposed to [d][key]) */
        #pragma unroll
        for (int e = tid; e < TILE_ELTS; e += 256) {
            const int ky = e >> 6;
            const int dd = e & 63;
            sKt[dd * KBLK + ky] = (__bf16)rawK[buf][e];
            sV[e]               = (__bf16)rawV[buf][e];
        }
        __syncthreads();

        /* wave-uniform dispatch: interior blocks run the mask-free fast path;
           exactly one diagonal block per wave takes the masked path. */
        if (kb < kb_diag) {
            attn_block<false>(kb, q0, half, ln, lane, qa, acc, rmax, rsum,
                              sKt, sV, &sP[wave * 16 * KBLK]);
        } else if (kb == kb_diag) {
            attn_block<true>(kb, q0, half, ln, lane, qa, acc, rmax, rsum,
                             sKt, sV, &sP[wave * 16 * KBLK]);
        }
        /* kb > kb_diag: fully masked for this wave -> skip */

        /* next tile must be fully landed in raw[buf^1] before next convert */
        if (ib + 1 < nblk) WAIT_ASYNC0();
        __syncthreads();
    }

    /* ---- epilogue: reduce the deferred per-lane denominators, then O = acc/sum */
    #pragma unroll
    for (int v = 0; v < 8; ++v) {
        const float total = row16_sum(rsum[v]);
        const float inv = 1.0f / total;
        float* orow = Op + (size_t)(q0 + v + 8 * half) * D_HEAD + ln;
        #pragma unroll
        for (int c = 0; c < 4; ++c) orow[16 * c] = acc[c][v] * inv;
    }
}

extern "C" void kernel_launch(void* const* d_in, const int* in_sizes, int n_in,
                              void* d_out, int out_size, void* d_ws, size_t ws_size,
                              hipStream_t stream) {
    (void)in_sizes; (void)n_in; (void)out_size; (void)d_ws; (void)ws_size;
    const float* Q = (const float*)d_in[0];
    const float* K = (const float*)d_in[1];
    const float* V = (const float*)d_in[2];
    /* d_in[3] is the boolean causal mask; causality is computed analytically. */
    float* O = (float*)d_out;

    const int BH = 2 * 16;
    dim3 grid(S_LEN / WG_ROWS, BH);   /* (16, 32) */
    dim3 block(256);
    fa_fwd_causal_bf16wmma<<<grid, block, 0, stream>>>(Q, K, V, O);
}
